// PairRewardSparseUNet_76244259438715
// MI455X (gfx1250) — compile-verified, matrix operand order checked by reading, not run on hardware
//
#include <hip/hip_runtime.h>

// ---------------------------------------------------------------------------
// Problem constants (fixed by the reference)
// ---------------------------------------------------------------------------
#define B_BATCH   16
#define PTS       50000
#define PP_PER_B  2048
#define C_FEAT    64
#define HID       1024
#define DIN       192                       // 3*C
#define NPAIR     (B_BATCH * PP_PER_B)      // 32768
#define HASH_BITS 21
#define HASH_SIZE (1u << HASH_BITS)
#define HASH_MASK (HASH_SIZE - 1u)
#define EMPTY_KEY 0xFFFFFFFFFFFFFFFFull

// ---------------------------------------------------------------------------
// CDNA5 feature detection: Tensor Data Mover (async tile DMA into LDS)
// ---------------------------------------------------------------------------
#if defined(__has_builtin)
#  if __has_builtin(__builtin_amdgcn_tensor_load_to_lds) && \
      __has_builtin(__builtin_amdgcn_s_wait_tensorcnt)
#    define HAVE_TDM 1
#  endif
#endif
#ifndef HAVE_TDM
#  define HAVE_TDM 0
#endif

// ---------------------------------------------------------------------------
// WMMA types (CDNA5 / gfx1250, wave32): 16x16x32 bf16 -> f32
// ---------------------------------------------------------------------------
typedef __attribute__((ext_vector_type(16))) __bf16 v16bf;
typedef __attribute__((ext_vector_type(8)))  float  v8f;
typedef __attribute__((ext_vector_type(4)))  unsigned v4u;
typedef __attribute__((ext_vector_type(8)))  int    v8i;
typedef __attribute__((ext_vector_type(4)))  int    v4i;

union Frag {
    v16bf v;
    uint4 q[2];   // 32 bytes = 8 VGPRs of bf16x2
};

static __device__ __forceinline__ unsigned short f32_to_bf16(float f) {
    unsigned u = __float_as_uint(f);
    u += 0x7FFFu + ((u >> 16) & 1u);        // round-to-nearest-even
    return (unsigned short)(u >> 16);
}

static __device__ __forceinline__ unsigned hash_slot(unsigned long long key) {
    return (unsigned)((key * 0x9E3779B97F4A7C15ull) >> (64 - HASH_BITS)) & HASH_MASK;
}

#if HAVE_TDM
#if __clang_major__ >= 23
#  define TDM_CALL(g0, g1, g2, g3) \
     __builtin_amdgcn_tensor_load_to_lds((g0), (g1), (g2), (g3), (v8i)0, 0)
#else
#  define TDM_CALL(g0, g1, g2, g3) \
     __builtin_amdgcn_tensor_load_to_lds((g0), (g1), (g2), (g3), 0)
#endif

// Low 32 bits of a shared-aperture flat address are the LDS byte offset
// (ISA 10.2).
static __device__ __forceinline__ unsigned lds_off(const void* p) {
    return (unsigned)(uintptr_t)p;
}

// 2-D tile DMA: tile0 elements per row (bf16), tile1 rows, row stride
// stride0 (elements).  D# bitfields per CDNA5 ISA 8.3/8.4.
static __device__ __forceinline__ void tdm_load_2d(
        unsigned lds_byte_off, const void* gptr,
        unsigned tile0, unsigned tile1,
        unsigned tdim0, unsigned tdim1,
        unsigned long long stride0) {
    unsigned long long ga = (unsigned long long)(uintptr_t)gptr;
    v4u g0;
    g0.x = 1u;                                              // count=1 (user D#)
    g0.y = lds_byte_off;                                    // lds_addr
    g0.z = (unsigned)ga;                                    // global_addr[31:0]
    g0.w = (unsigned)((ga >> 32) & 0x01FFFFFFu) | (2u << 30); // addr[56:32]|type=2
    v8i g1;
    g1[0] = (int)(1u << 16);                                // data_size=1 (2B)
    g1[1] = (int)((tdim0 & 0xFFFFu) << 16);                 // tensor_dim0[15:0]
    g1[2] = (int)((tdim0 >> 16) | ((tdim1 & 0xFFFFu) << 16));
    g1[3] = (int)((tdim1 >> 16) | (tile0 << 16));           // tile_dim0
    g1[4] = (int)(tile1 & 0xFFFFu);                         // tile_dim1, dim2=0
    g1[5] = (int)(unsigned)(stride0 & 0xFFFFFFFFull);       // dim0_stride[31:0]
    g1[6] = (int)(unsigned)((stride0 >> 32) & 0xFFFFull);   // dim0_stride[47:32]
    g1[7] = 0;
    v4i z = (v4i)0;
    TDM_CALL(g0, g1, z, z);
}
#endif // HAVE_TDM

// ---------------------------------------------------------------------------
// Workspace init (hash table, pred accumulator, per-batch stats)
// ---------------------------------------------------------------------------
__global__ void init_ws(unsigned long long* __restrict__ hk,
                        unsigned* __restrict__ hv,
                        float* __restrict__ pred,
                        float* __restrict__ stats) {
    unsigned i = blockIdx.x * blockDim.x + threadIdx.x;
    if (i < HASH_SIZE) { hk[i] = EMPTY_KEY; hv[i] = 0xFFFFFFFFu; }
    if (i < NPAIR)     pred[i]  = 0.0f;
    if (i < 64)        stats[i] = 0.0f;
}

// ---------------------------------------------------------------------------
// Hash insert: key = pack16(batch,x,y,z); duplicate keys keep the MIN row
// index (matches stable argsort + searchsorted-left in the reference).
// ---------------------------------------------------------------------------
__global__ void hash_insert(const long long* __restrict__ vidx,
                            unsigned long long* __restrict__ hk,
                            unsigned* __restrict__ hv, int N) {
    int i = blockIdx.x * blockDim.x + threadIdx.x;
    if (i >= N) return;
    const long long* r = vidx + (size_t)i * 4;
    unsigned long long key =
        ((unsigned long long)(r[0] & 0xFFFF) << 48) |
        ((unsigned long long)(r[1] & 0xFFFF) << 32) |
        ((unsigned long long)(r[2] & 0xFFFF) << 16) |
        ((unsigned long long)(r[3] & 0xFFFF));
    unsigned slot = hash_slot(key);
    for (;;) {
        unsigned long long old = atomicCAS(&hk[slot], EMPTY_KEY, key);
        if (old == EMPTY_KEY || old == key) {
            atomicMin(&hv[slot], (unsigned)i);
            break;
        }
        slot = (slot + 1) & HASH_MASK;
    }
}

static __device__ __forceinline__ unsigned hash_lookup(
        const unsigned long long* __restrict__ hk,
        const unsigned* __restrict__ hv, unsigned long long key) {
    unsigned slot = hash_slot(key);
    for (unsigned probe = 0; probe < HASH_SIZE; ++probe) {
        unsigned long long k = hk[slot];
        if (k == key)       return hv[slot];
        if (k == EMPTY_KEY) return 0u;      // cannot happen by construction
        slot = (slot + 1) & HASH_MASK;
    }
    return 0u;
}

// ---------------------------------------------------------------------------
// Wave-per-pair gather: lookup both endpoints, emit [f0 | f1 | f0-f1] as bf16
// row-major (32768 x 192).
// ---------------------------------------------------------------------------
__global__ __launch_bounds__(256) void gather_pairs(
        const float* __restrict__ vf, const int* __restrict__ grid,
        const long long* __restrict__ pairs,
        const unsigned long long* __restrict__ hk,
        const unsigned* __restrict__ hv,
        unsigned short* __restrict__ Xp) {
    const int wid  = threadIdx.x >> 5;
    const int lane = threadIdx.x & 31;
    const int p    = blockIdx.x * 8 + wid;
    const int seg  = p >> 11;               // p / PP_PER_B
    unsigned r = 0;
    if (lane < 2) {
        long long i = pairs[(size_t)2 * p + lane] + (long long)seg * PTS;
        int b = (int)(i / PTS);             // == searchsorted(offset, i, right)
        const int* g = grid + (size_t)i * 3;
        unsigned long long key =
            ((unsigned long long)(b    & 0xFFFF) << 48) |
            ((unsigned long long)(g[0] & 0xFFFF) << 32) |
            ((unsigned long long)(g[1] & 0xFFFF) << 16) |
            ((unsigned long long)(g[2] & 0xFFFF));
        r = hash_lookup(hk, hv, key);
    }
    unsigned r0 = (unsigned)__shfl((int)r, 0);
    unsigned r1 = (unsigned)__shfl((int)r, 1);
    unsigned short* out = Xp + (size_t)p * DIN;
    for (int c = lane; c < C_FEAT; c += 32) {
        float a0 = vf[(size_t)r0 * C_FEAT + c];
        float a1 = vf[(size_t)r1 * C_FEAT + c];
        out[c]              = f32_to_bf16(a0);
        out[C_FEAT + c]     = f32_to_bf16(a1);
        out[2 * C_FEAT + c] = f32_to_bf16(a0 - a1);
    }
}

// ---------------------------------------------------------------------------
// Pack a K x N fp32 weight matrix into the per-lane WMMA B-fragment layout:
// tile (kt, nt) is 32x16; lane l holds column nt*16 + (l&15), with
// K chunks {0..7,16..23} (l<16) or {8..15,24..31} (l>=16) contiguously,
// so each lane's fragment is one contiguous 32-byte run, and one block's
// eight B tiles per K-step form a single contiguous 8 KB region (TDM-able).
// ---------------------------------------------------------------------------
__global__ void pack_w(const float* __restrict__ W, unsigned short* __restrict__ Wp,
                       int K, int N) {
    int idx = blockIdx.x * blockDim.x + threadIdx.x;
    if (idx >= K * N) return;
    int e    = idx & 15;
    int lane = (idx >> 4) & 31;
    int tile = idx >> 9;                    // 512 elements per 32x16 tile
    int ntiles = N >> 4;
    int kt = tile / ntiles, nt = tile % ntiles;
    int kbase = (lane < 16) ? 0 : 8;
    int k = kt * 32 + kbase + (e < 8 ? e : e + 8);
    int n = nt * 16 + (lane & 15);
    Wp[idx] = f32_to_bf16(W[(size_t)k * N + n]);
}

// ---------------------------------------------------------------------------
// GEMM main loops.  TDM path: double-buffered LDS tiles filled by the Tensor
// Data Mover (wave 0 issues, TENSORcnt synchronizes), fragments read with
// ds_load_b128.  Fallback path: direct global_load_b128 per wave.
// ---------------------------------------------------------------------------
#define GEMM_PREAMBLE                                                   \
    const int lane  = threadIdx.x & 31;                                 \
    const int wid   = threadIdx.x >> 5;                                 \
    const int wm    = wid & 1, wn = wid >> 1;                           \
    const int bm    = blockIdx.x * 128, bn = blockIdx.y * 128;          \
    const int half8 = (lane >> 4) << 3;     /* 0 or 8 */                \
    const int ntiles = N >> 4;                                          \
    v8f acc[4][2] = {};

#if HAVE_TDM
#define GEMM_MAINLOOP(A_, Bp_, M_, N_, K_)                                   \
    __shared__ unsigned short As[2][128 * 32];                               \
    __shared__ unsigned short Bs[2][8 * 512];                                \
    const int nk = (K_) >> 5;                                                \
    if (threadIdx.x < 32) {                                                  \
        tdm_load_2d(lds_off(As[0]), (A_) + (size_t)bm * (K_), 32, 128,       \
                    (unsigned)(K_), (unsigned)(M_), (unsigned long long)(K_)); \
        tdm_load_2d(lds_off(Bs[0]), (Bp_) + (size_t)(bn >> 4) * 512,         \
                    4096, 1, 0x7FFFFFFFu, 1, 4096ull);                       \
    }                                                                        \
    for (int kt = 0; kt < nk; ++kt) {                                        \
        const int cur = kt & 1;                                              \
        if (threadIdx.x < 32) __builtin_amdgcn_s_wait_tensorcnt(0);          \
        __syncthreads();                                                     \
        if (threadIdx.x < 32 && kt + 1 < nk) {                               \
            tdm_load_2d(lds_off(As[cur ^ 1]),                                \
                        (A_) + (size_t)bm * (K_) + (kt + 1) * 32, 32, 128,   \
                        (unsigned)(K_), (unsigned)(M_),                      \
                        (unsigned long long)(K_));                           \
            tdm_load_2d(lds_off(Bs[cur ^ 1]),                                \
                        (Bp_) + ((size_t)(kt + 1) * ntiles + (bn >> 4)) * 512,\
                        4096, 1, 0x7FFFFFFFu, 1, 4096ull);                   \
        }                                                                    \
        Frag a[4], b[2];                                                     \
        _Pragma("unroll")                                                    \
        for (int mi = 0; mi < 4; ++mi) {                                     \
            const unsigned short* p =                                        \
                As[cur] + (wm * 64 + mi * 16 + (lane & 15)) * 32 + half8;    \
            a[mi].q[0] = *(const uint4*)p;                                   \
            a[mi].q[1] = *(const uint4*)(p + 16);                            \
        }                                                                    \
        _Pragma("unroll")                                                    \
        for (int ni = 0; ni < 2; ++ni) {                                     \
            const unsigned short* p =                                        \
                Bs[cur] + (wn * 2 + ni) * 512 + lane * 16;                   \
            b[ni].q[0] = *(const uint4*)p;                                   \
            b[ni].q[1] = *(const uint4*)(p + 8);                             \
        }                                                                    \
        _Pragma("unroll")                                                    \
        for (int mi = 0; mi < 4; ++mi)                                       \
            _Pragma("unroll")                                                \
            for (int ni = 0; ni < 2; ++ni)                                   \
                acc[mi][ni] = __builtin_amdgcn_wmma_f32_16x16x32_bf16(       \
                    false, a[mi].v, false, b[ni].v, (short)0, acc[mi][ni],   \
                    false, false);                                           \
        __syncthreads();                                                     \
    }
#else
#define GEMM_MAINLOOP(A_, Bp_, M_, N_, K_)                                   \
    const int rowA = bm + wm * 64 + (lane & 15);                             \
    for (int k0 = 0; k0 < (K_); k0 += 32) {                                  \
        Frag a[4], b[2];                                                     \
        _Pragma("unroll")                                                    \
        for (int mi = 0; mi < 4; ++mi) {                                     \
            const unsigned short* p =                                        \
                (A_) + (size_t)(rowA + mi * 16) * (K_) + k0 + half8;         \
            a[mi].q[0] = *(const uint4*)p;                                   \
            a[mi].q[1] = *(const uint4*)(p + 16);                            \
        }                                                                    \
        _Pragma("unroll")                                                    \
        for (int ni = 0; ni < 2; ++ni) {                                     \
            const int nt = (bn >> 4) + wn * 2 + ni;                          \
            const unsigned short* p =                                        \
                (Bp_) + (((size_t)(k0 >> 5) * ntiles + nt) * 32 + lane) * 16;\
            b[ni].q[0] = *(const uint4*)p;                                   \
            b[ni].q[1] = *(const uint4*)(p + 8);                             \
        }                                                                    \
        if (k0 + 32 < (K_))                                                  \
            __builtin_prefetch((A_) + (size_t)rowA * (K_) + k0 + 32, 0, 1);  \
        _Pragma("unroll")                                                    \
        for (int mi = 0; mi < 4; ++mi)                                       \
            _Pragma("unroll")                                                \
            for (int ni = 0; ni < 2; ++ni)                                   \
                acc[mi][ni] = __builtin_amdgcn_wmma_f32_16x16x32_bf16(       \
                    false, a[mi].v, false, b[ni].v, (short)0, acc[mi][ni],   \
                    false, false);                                           \
    }
#endif

// GEMM1: (M x K bf16 row-major) @ packed(K x N) + bias -> relu -> bf16 out.
__global__ __launch_bounds__(256) void gemm_bias_relu(
        const unsigned short* __restrict__ A, const unsigned short* __restrict__ Bp,
        const float* __restrict__ bias, unsigned short* __restrict__ Cout,
        int M, int N, int K) {
    GEMM_PREAMBLE
    GEMM_MAINLOOP(A, Bp, M, N, K)
#pragma unroll
    for (int mi = 0; mi < 4; ++mi)
#pragma unroll
        for (int ni = 0; ni < 2; ++ni) {
            const int col = bn + wn * 32 + ni * 16 + (lane & 15);
            const float bv = bias[col];
#pragma unroll
            for (int v = 0; v < 8; ++v) {
                const int row = bm + wm * 64 + mi * 16 + half8 + v;
                float x = fmaxf(acc[mi][ni][v] + bv, 0.0f);
                Cout[(size_t)row * N + col] = f32_to_bf16(x);
            }
        }
}

// GEMM2 fused with the final 1024->1 projection: relu(acc + b2) * W3[col],
// shfl_xor reduction across the 16 columns each lane-half holds, then one
// atomicAdd per row per wave (4 per N-block).  H2 is never materialized.
__global__ __launch_bounds__(256) void gemm_relu_dotw3(
        const unsigned short* __restrict__ A, const unsigned short* __restrict__ Bp,
        const float* __restrict__ bias, const float* __restrict__ W3,
        float* __restrict__ pred, int M, int N, int K) {
    GEMM_PREAMBLE
    GEMM_MAINLOOP(A, Bp, M, N, K)
    float rsum[4][8] = {};
#pragma unroll
    for (int mi = 0; mi < 4; ++mi)
#pragma unroll
        for (int ni = 0; ni < 2; ++ni) {
            const int col = bn + wn * 32 + ni * 16 + (lane & 15);
            const float bv = bias[col];
            const float wv = W3[col];
#pragma unroll
            for (int v = 0; v < 8; ++v) {
                float h = fmaxf(acc[mi][ni][v] + bv, 0.0f);
                rsum[mi][v] += h * wv;
            }
        }
#pragma unroll
    for (int mi = 0; mi < 4; ++mi)
#pragma unroll
        for (int v = 0; v < 8; ++v) {
            float s = rsum[mi][v];
            s += __shfl_xor(s, 1);
            s += __shfl_xor(s, 2);
            s += __shfl_xor(s, 4);
            s += __shfl_xor(s, 8);          // lanes 0 / 16 now hold row sums
            if ((lane & 15) == 0) {
                const int row = bm + wm * 64 + mi * 16 + half8 + v;
                atomicAdd(&pred[row], s);
            }
        }
}

// ---------------------------------------------------------------------------
// Per-batch ranking + MSE statistics.  One block per batch; preds/targets in
// LDS; numerically stable softplus.
// ---------------------------------------------------------------------------
__global__ __launch_bounds__(256) void rank_stats(
        const float* __restrict__ pred, const float* __restrict__ reward,
        const float* __restrict__ b3, float* __restrict__ stats) {
    __shared__ float sp[PP_PER_B];
    __shared__ float st[PP_PER_B];
    __shared__ float rs[256], rm[256];
    __shared__ unsigned rc[256];
    const int b = blockIdx.x, tid = threadIdx.x;
    const float bias3 = b3[0];
    for (int i = tid; i < PP_PER_B; i += 256) {
        sp[i] = pred[b * PP_PER_B + i] + bias3;
        st[i] = reward[b * PP_PER_B + i];
    }
    __syncthreads();
    float sum = 0.0f, mse = 0.0f;
    unsigned cnt = 0;
    for (int t = tid; t < PP_PER_B * PP_PER_B; t += 256) {
        int i = t >> 11, j = t & (PP_PER_B - 1);
        if (j > i) {
            float dt = st[i] - st[j];
            if (dt != 0.0f) {
                float sgn = dt > 0.0f ? 1.0f : -1.0f;
                float x = -sgn * (sp[i] - sp[j]);
                sum += fmaxf(x, 0.0f) + log1pf(expf(-fabsf(x)));
                ++cnt;
            }
        }
    }
    for (int i = tid; i < PP_PER_B; i += 256) {
        float d = sp[i] - st[i];
        mse += d * d;
    }
    rs[tid] = sum; rm[tid] = mse; rc[tid] = cnt;
    __syncthreads();
    for (int s = 128; s > 0; s >>= 1) {
        if (tid < s) { rs[tid] += rs[tid + s]; rm[tid] += rm[tid + s]; rc[tid] += rc[tid + s]; }
        __syncthreads();
    }
    if (tid == 0) {
        stats[b * 4 + 0] = rs[0];
        stats[b * 4 + 1] = (float)rc[0];
        stats[b * 4 + 2] = rm[0];
    }
}

__global__ void final_loss(const float* __restrict__ stats, float* __restrict__ out) {
    float rl = 0.0f, ml = 0.0f;
    int nh = 0;
    for (int b = 0; b < B_BATCH; ++b) {
        float sum = stats[b * 4 + 0];
        float cnt = stats[b * 4 + 1];
        float mse = stats[b * 4 + 2];
        if (cnt > 0.0f) { rl += sum / cnt; ++nh; }
        ml += mse / (float)PP_PER_B;
    }
    rl = rl / (float)(nh > 0 ? nh : 1);
    out[0] = rl + 0.1f * (ml / (float)B_BATCH);
}

// ---------------------------------------------------------------------------
// Host launcher
// ---------------------------------------------------------------------------
extern "C" void kernel_launch(void* const* d_in, const int* in_sizes, int n_in,
                              void* d_out, int out_size, void* d_ws, size_t ws_size,
                              hipStream_t stream) {
    const float* voxel_features = (const float*)d_in[0];
    const float* pair_reward    = (const float*)d_in[1];
    const float* W1 = (const float*)d_in[2];
    const float* b1 = (const float*)d_in[3];
    const float* W2 = (const float*)d_in[4];
    const float* b2 = (const float*)d_in[5];
    const float* W3 = (const float*)d_in[6];
    const float* b3 = (const float*)d_in[7];
    const int*       grid_coord    = (const int*)d_in[8];
    const long long* voxel_indices = (const long long*)d_in[10];
    const long long* pairs         = (const long long*)d_in[11];

    const int Nvox = in_sizes[10] / 4;      // 800000
    const int P    = in_sizes[11] / 2;      // 32768

    char* ws = (char*)d_ws;
    size_t off = 0;
    auto take = [&](size_t bytes) -> char* {
        char* p = ws + off;
        off = (off + bytes + 255) & ~(size_t)255;
        return p;
    };
    unsigned long long* hkeys = (unsigned long long*)take((size_t)HASH_SIZE * 8);
    unsigned*           hvals = (unsigned*)take((size_t)HASH_SIZE * 4);
    unsigned short*     W1p   = (unsigned short*)take((size_t)DIN * HID * 2);
    unsigned short*     W2p   = (unsigned short*)take((size_t)HID * HID * 2);
    unsigned short*     Xp    = (unsigned short*)take((size_t)NPAIR * DIN * 2);
    unsigned short*     H1    = (unsigned short*)take((size_t)NPAIR * HID * 2);
    float*              pred  = (float*)take((size_t)NPAIR * 4);
    float*              stats = (float*)take(256);

    init_ws<<<(HASH_SIZE + 255) / 256, 256, 0, stream>>>(hkeys, hvals, pred, stats);
    hash_insert<<<(Nvox + 255) / 256, 256, 0, stream>>>(voxel_indices, hkeys, hvals, Nvox);
    pack_w<<<(DIN * HID + 255) / 256, 256, 0, stream>>>(W1, W1p, DIN, HID);
    pack_w<<<(HID * HID + 255) / 256, 256, 0, stream>>>(W2, W2p, HID, HID);
    gather_pairs<<<P / 8, 256, 0, stream>>>(voxel_features, grid_coord, pairs,
                                            hkeys, hvals, Xp);
    gemm_bias_relu<<<dim3(NPAIR / 128, HID / 128), 256, 0, stream>>>(
        Xp, W1p, b1, H1, NPAIR, HID, DIN);
    gemm_relu_dotw3<<<dim3(NPAIR / 128, HID / 128), 256, 0, stream>>>(
        H1, W2p, b2, W3, pred, NPAIR, HID, HID);
    rank_stats<<<B_BATCH, 256, 0, stream>>>(pred, pair_reward, b3, stats);
    final_loss<<<1, 1, 0, stream>>>(stats, (float*)d_out);
}
